// GraphEncoder_74749610819667
// MI455X (gfx1250) — matrix operand order repaired and checked by reading, not compiled, not run on hardware
//
#include <hip/hip_runtime.h>
#include <hip/hip_bf16.h>
#include <stdint.h>

// CDNA5 WMMA fragment types (wave32)
typedef __attribute__((ext_vector_type(16))) __bf16   v16bf;
typedef __attribute__((ext_vector_type(8)))  float    v8f;
typedef __attribute__((ext_vector_type(4)))  unsigned uv4;

union FragBF { v16bf v; uv4 q[2]; };

__device__ __forceinline__ unsigned f2bf(float f) {
  unsigned x = __builtin_bit_cast(unsigned, f);
  return (x + 0x7FFFu + ((x >> 16) & 1u)) >> 16;   // round-to-nearest-even
}

// ---------------- elementwise / prep kernels ----------------

__global__ void k_fill(float* __restrict__ p, float v, int n) {
  int i = blockIdx.x * blockDim.x + threadIdx.x;
  if (i < n) p[i] = v;
}

__global__ void k_degree(const int* __restrict__ dst, float* __restrict__ deg, int E) {
  int e = blockIdx.x * blockDim.x + threadIdx.x;
  if (e < E) atomicAdd(&deg[dst[e]], 1.0f);
}

__global__ void k_rsqrt(const float* __restrict__ deg, float* __restrict__ dis, int n) {
  int i = blockIdx.x * blockDim.x + threadIdx.x;
  if (i < n) dis[i] = rsqrtf(deg[i]);
}

// int32 src/dst + fused norm; entries [E, E+N) are the self-loops
__global__ void k_edgeprep(const int* __restrict__ ei, const float* __restrict__ dis,
                           int* __restrict__ s32, int* __restrict__ d32,
                           float* __restrict__ nrm, int E, int N) {
  int e = blockIdx.x * blockDim.x + threadIdx.x;
  if (e >= E + N) return;
  if (e < E) {
    int s = ei[e], d = ei[E + e];
    s32[e] = s; d32[e] = d;
    nrm[e] = dis[s] * dis[d];
  } else {
    int i = e - E;
    s32[e] = i; d32[e] = i;
    float di = dis[i];
    nrm[e] = di * di;
  }
}

__global__ void k_f32_to_bf16(const float* __restrict__ in,
                              unsigned short* __restrict__ out, int n) {
  int i = blockIdx.x * blockDim.x + threadIdx.x;
  if (i < n) out[i] = (unsigned short)f2bf(in[i]);
}

// Pack W[K][Ncols] (f32 row-major) into CDNA5 16-bit B-fragment layout:
// fragment id = ct*KC + kc; per lane l (col n = ct*16 + l%16, half h = l/16),
// word v packs K = kc*32 + 16(v>>2) + 8h + 2(v&3) and K+1. Stored lane-major: Wp[(frag*32+l)*8+v]
__global__ void k_pack_w(const float* __restrict__ W, unsigned* __restrict__ Wp,
                         int K, int Ncols) {
  int KC = K >> 5;
  int total = (Ncols >> 4) * KC * 256;
  int tid = blockIdx.x * blockDim.x + threadIdx.x;
  if (tid >= total) return;
  int v    = tid & 7;
  int lane = (tid >> 3) & 31;
  int frag = tid >> 8;
  int kc = frag % KC, ct = frag / KC;
  int n = ct * 16 + (lane & 15);
  int h = lane >> 4;
  int k = kc * 32 + 16 * (v >> 2) + 8 * h + 2 * (v & 3);
  unsigned lo = f2bf(W[(size_t)k * Ncols + n]);
  unsigned hi = f2bf(W[(size_t)(k + 1) * Ncols + n]);
  Wp[tid] = lo | (hi << 16);
}

// ---------------- WMMA GEMM: out[M,Ncols] = A[M,K](bf16) @ Wpacked ----------------
// 4 waves per block; each wave computes a 16 x (16*CT) output strip, loading the
// A fragment once per K-chunk and reusing it across CT column tiles.
// K multiple of 32; Ncols multiple of 16*CT.
template <int CT>
__global__ void __launch_bounds__(128) k_gemm_wmma(
    const unsigned short* __restrict__ A, const unsigned* __restrict__ Wp,
    float* __restrict__ out, const float* __restrict__ bias,
    int M, int K, int Ncols, int relu) {
  const int wave = threadIdx.x >> 5;
  const int l = threadIdx.x & 31;
  const int rt = blockIdx.x * 4 + wave;            // row tile
  if (rt * 16 >= M) return;
  const int ctbase = blockIdx.y * CT;              // first column tile
  const int m = l & 15, h = l >> 4;
  const int KC = K >> 5;

  int rowA = rt * 16 + m;
  if (rowA >= M) rowA = M - 1;
  const uv4* arow = (const uv4*)(A + (size_t)rowA * K);   // K/8 uv4 per row

  v8f c[CT];
#pragma unroll
  for (int t = 0; t < CT; ++t) c[t] = {};

  for (int kc = 0; kc < KC; ++kc) {
    // A (16-bit, 16x32): VGPR0-3 <- K = kc*32 + 8h + {0..7}; VGPR4-7 <- +16
    FragBF a;
    a.q[0] = arow[kc * 4 + h];
    a.q[1] = arow[kc * 4 + h + 2];
#pragma unroll
    for (int t = 0; t < CT; ++t) {
      FragBF b;
      const uv4* wp =
          (const uv4*)(Wp + ((size_t)((ctbase + t) * KC + kc) * 32 + (size_t)l) * 8);
      b.q[0] = wp[0];
      b.q[1] = wp[1];
      c[t] = __builtin_amdgcn_wmma_f32_16x16x32_bf16(
          /*neg_a=*/false, a.v, /*neg_b=*/false, b.v,
          /*c_mod=*/(short)0, c[t], /*reuse_a=*/false, /*reuse_b=*/false);
    }
  }

  const int rowbase = rt * 16 + 8 * h;             // D: VGPR r -> M = r + 8h
#pragma unroll
  for (int t = 0; t < CT; ++t) {
    const int col = (ctbase + t) * 16 + m;
    const float bv = bias ? bias[col] : 0.0f;
    float* o = out + (size_t)rowbase * Ncols + col;
#pragma unroll
    for (int r = 0; r < 8; ++r) {
      if (rowbase + r < M) {
        float val = c[t][r] + bv;
        if (relu) val = fmaxf(val, 0.0f);
        o[(size_t)r * Ncols] = val;
      }
    }
  }
}

// ---------------- aggregation / pooling ----------------

__global__ void k_bias_init(float* __restrict__ out, const float* __restrict__ b, int total) {
  int i = blockIdx.x * blockDim.x + threadIdx.x;
  if (i < total) out[i] = b[i & 63];
}

// out[dst] += norm * t[src]; 16 threads per edge, float4 each (H=64)
__global__ void k_aggregate(const float* __restrict__ t, const int* __restrict__ s32,
                            const int* __restrict__ d32, const float* __restrict__ nrm,
                            float* __restrict__ out, int E2) {
  unsigned idx = blockIdx.x * blockDim.x + threadIdx.x;
  unsigned e = idx >> 4;
  if (e >= (unsigned)E2) return;
  int fq = (int)(idx & 15u) << 2;
  int s = s32[e], d = d32[e];
  float w = nrm[e];
  const float4 v = *(const float4*)(t + (size_t)s * 64 + fq);
  float* o = out + (size_t)d * 64 + fq;
  atomicAdd(o + 0, w * v.x);
  atomicAdd(o + 1, w * v.y);
  atomicAdd(o + 2, w * v.z);
  atomicAdd(o + 3, w * v.w);
}

__global__ void k_relu(float* __restrict__ p, int n) {
  int i = blockIdx.x * blockDim.x + threadIdx.x;
  if (i < n) p[i] = fmaxf(p[i], 0.0f);
}

__global__ void k_pool_count(const int* __restrict__ batch, float* __restrict__ cnt, int N) {
  int i = blockIdx.x * blockDim.x + threadIdx.x;
  if (i < N) atomicAdd(&cnt[batch[i]], 1.0f);
}

__global__ void k_pool_sum(const float* __restrict__ h, const int* __restrict__ batch,
                           float* __restrict__ pooled, int N) {
  int idx = blockIdx.x * blockDim.x + threadIdx.x;
  if (idx >= N * 64) return;
  int i = idx >> 6, f = idx & 63;
  atomicAdd(&pooled[(size_t)batch[i] * 64 + f], h[idx]);
}

__global__ void k_pool_div(float* __restrict__ pooled, const float* __restrict__ cnt, int total) {
  int idx = blockIdx.x * blockDim.x + threadIdx.x;
  if (idx < total) pooled[idx] /= fmaxf(cnt[idx >> 6], 1.0f);
}

// ---------------- launch ----------------

extern "C" void kernel_launch(void* const* d_in, const int* in_sizes, int n_in,
                              void* d_out, int out_size, void* d_ws, size_t ws_size,
                              hipStream_t stream) {
  (void)n_in; (void)ws_size;
  const float* x   = (const float*)d_in[0];
  const int*   ei  = (const int*)d_in[1];
  const int*   bat = (const int*)d_in[2];
  const float* W[5] = { (const float*)d_in[3], (const float*)d_in[5], (const float*)d_in[7],
                        (const float*)d_in[9], (const float*)d_in[11] };
  const float* b[5] = { (const float*)d_in[4], (const float*)d_in[6], (const float*)d_in[8],
                        (const float*)d_in[10], (const float*)d_in[12] };
  const float* Wh1 = (const float*)d_in[13];
  const float* bh1 = (const float*)d_in[14];
  const float* Wh2 = (const float*)d_in[15];
  const float* bh2 = (const float*)d_in[16];

  const int N    = in_sizes[2];
  const int E    = in_sizes[1] / 2;
  const int F    = in_sizes[0] / N;   // 64
  const int H    = in_sizes[4];       // 64
  const int NHID = in_sizes[14];      // 256
  const int NOUT = in_sizes[16];      // 128
  const int G    = out_size / NOUT;   // 2000
  const int E2   = E + N;

  // workspace carve
  char* wsp = (char*)d_ws;
  auto alloc = [&](size_t bytes) -> char* {
    char* p = wsp;
    wsp += (bytes + 255) & ~(size_t)255;
    return p;
  };
  float* deg = (float*)alloc((size_t)N * 4);
  float* dis = (float*)alloc((size_t)N * 4);
  int*   s32 = (int*)alloc((size_t)E2 * 4);
  int*   d32 = (int*)alloc((size_t)E2 * 4);
  float* nrm = (float*)alloc((size_t)E2 * 4);
  unsigned short* hb = (unsigned short*)alloc((size_t)N * H * 2);  // also reused for MLP acts
  float* tbuf = (float*)alloc((size_t)N * H * 4);
  float* hA   = (float*)alloc((size_t)N * H * 4);
  float* hB   = (float*)alloc((size_t)N * H * 4);
  unsigned* WpG[5];
  for (int i = 0; i < 5; ++i) {
    int K = (i == 0) ? F : H;
    WpG[i] = (unsigned*)alloc((size_t)(H / 16) * (K / 32) * 256 * 4);
  }
  unsigned* WpH1 = (unsigned*)alloc((size_t)(NHID / 16) * (H / 32) * 256 * 4);
  unsigned* WpH2 = (unsigned*)alloc((size_t)(NOUT / 16) * (NHID / 32) * 256 * 4);
  float* pooled = (float*)alloc((size_t)G * H * 4);
  float* cnt    = (float*)alloc((size_t)G * 4);
  float* z      = (float*)alloc((size_t)G * NHID * 4);

  const int TB = 256;
  auto gr = [](long long n, int blk) { return (unsigned)((n + blk - 1) / blk); };

  // degree + normalization (self-loops folded as deg init = 1)
  k_fill<<<gr(N, TB), TB, 0, stream>>>(deg, 1.0f, N);
  k_degree<<<gr(E, TB), TB, 0, stream>>>(ei + E, deg, E);
  k_rsqrt<<<gr(N, TB), TB, 0, stream>>>(deg, dis, N);
  k_edgeprep<<<gr(E2, TB), TB, 0, stream>>>(ei, dis, s32, d32, nrm, E, N);

  // pack all weights into WMMA B-fragment layout (once per launch)
  for (int i = 0; i < 5; ++i) {
    int K = (i == 0) ? F : H;
    k_pack_w<<<gr((long long)(H / 16) * (K / 32) * 256, TB), TB, 0, stream>>>(W[i], WpG[i], K, H);
  }
  k_pack_w<<<gr((long long)(NHID / 16) * (H / 32) * 256, TB), TB, 0, stream>>>(Wh1, WpH1, H, NHID);
  k_pack_w<<<gr((long long)(NOUT / 16) * (NHID / 32) * 256, TB), TB, 0, stream>>>(Wh2, WpH2, NHID, NOUT);

  // 5 GCN layers
  const float* hin = x;
  float* bufs[2] = { hA, hB };
  for (int layer = 0; layer < 5; ++layer) {
    int K = (layer == 0) ? F : H;
    float* hout = bufs[layer & 1];
    k_f32_to_bf16<<<gr((long long)N * K, TB), TB, 0, stream>>>(hin, hb, N * K);
    dim3 gg(gr(gr(N, 16), 4), H / 64);             // 4 row-tiles/block, 4 col-tiles/wave
    k_gemm_wmma<4><<<gg, 128, 0, stream>>>(hb, WpG[layer], tbuf, nullptr, N, K, H, 0);
    k_bias_init<<<gr((long long)N * H, TB), TB, 0, stream>>>(hout, b[layer], N * H);
    k_aggregate<<<gr((long long)E2 * 16, TB), TB, 0, stream>>>(tbuf, s32, d32, nrm, hout, E2);
    if (layer < 4) k_relu<<<gr((long long)N * H, TB), TB, 0, stream>>>(hout, N * H);
    hin = hout;
  }

  // global mean pool
  k_fill<<<gr((long long)G * H, TB), TB, 0, stream>>>(pooled, 0.0f, G * H);
  k_fill<<<gr(G, TB), TB, 0, stream>>>(cnt, 0.0f, G);
  k_pool_count<<<gr(N, TB), TB, 0, stream>>>(bat, cnt, N);
  k_pool_sum<<<gr((long long)N * H, TB), TB, 0, stream>>>(hin, bat, pooled, N);
  k_pool_div<<<gr((long long)G * H, TB), TB, 0, stream>>>(pooled, cnt, G * H);

  // MLP head (WMMA, fused bias + relu epilogue)
  k_f32_to_bf16<<<gr((long long)G * H, TB), TB, 0, stream>>>(pooled, hb, G * H);
  dim3 g1(gr(gr(G, 16), 4), NHID / 64);
  k_gemm_wmma<4><<<g1, 128, 0, stream>>>(hb, WpH1, z, bh1, G, H, NHID, 1);
  k_f32_to_bf16<<<gr((long long)G * NHID, TB), TB, 0, stream>>>(z, hb, G * NHID);
  dim3 g2(gr(gr(G, 16), 4), NOUT / 64);
  k_gemm_wmma<4><<<g2, 128, 0, stream>>>(hb, WpH2, (float*)d_out, bh2, G, NHID, NOUT, 0);
}
